// SpMiddleParallelPillarEncoder34HA_16432544875402
// MI455X (gfx1250) — compile-verified, matrix operand order checked by reading, not run on hardware
//
#include <hip/hip_runtime.h>
#include <cstdint>
#include <cstddef>

// ---------------------------------------------------------------------------
// Types for CDNA5 WMMA (wave32): v_wmma_f32_16x16x32_bf16
// ---------------------------------------------------------------------------
typedef __bf16 bf16;
typedef __attribute__((ext_vector_type(16))) __bf16 v16bf;
typedef __attribute__((ext_vector_type(8)))  __bf16 v8bf;
typedef __attribute__((ext_vector_type(8)))  float  v8f;

#define PC_MIN (-75.2f)

// ---------------------------------------------------------------------------
// Utility kernels
// ---------------------------------------------------------------------------
__global__ void fill_zero_kernel(float* __restrict__ p, size_t n) {
    size_t i  = (size_t)blockIdx.x * blockDim.x + threadIdx.x;
    size_t st = (size_t)gridDim.x * blockDim.x;
    for (; i < n; i += st) p[i] = 0.0f;
}

__global__ void cvt_f32_bf16_kernel(const float* __restrict__ a, bf16* __restrict__ o, size_t n) {
    size_t i  = (size_t)blockIdx.x * blockDim.x + threadIdx.x;
    size_t st = (size_t)gridDim.x * blockDim.x;
    for (; i < n; i += st) o[i] = (bf16)a[i];
}

// OIHW f32 -> [tap][cout][cin] bf16 (tap = ky*3+kx): a lane's K fragment
// (contiguous ci) becomes two 16B vector loads.
__global__ void repack_w_kernel(const float* __restrict__ w, bf16* __restrict__ o,
                                int Cin, int Cout, int total) {
    int t = blockIdx.x * blockDim.x + threadIdx.x;
    if (t >= total) return;
    int ci  = t % Cin;
    int r   = t / Cin;
    int co  = r % Cout;
    int tap = r / Cout;
    o[t] = (bf16)w[((size_t)co * Cin + ci) * 9 + tap];
}

// ---------------------------------------------------------------------------
// Pillar encoder: per-point MLP (8 -> Cout), ReLU, scatter-max into NHWC grid.
// ReLU output >= 0 so uint compare of float bits is order-preserving; pooled
// buffer is zero-initialized which matches "empty pillar -> 0".
// ---------------------------------------------------------------------------
__global__ void pillar_encode_kernel(const float* __restrict__ xyz,
                                     const float* __restrict__ ptf,
                                     const float* __restrict__ w,     // [8][Cout]
                                     const float* __restrict__ bias,  // [Cout]
                                     const float* __restrict__ g,
                                     const float* __restrict__ beta,
                                     unsigned int* __restrict__ pooled, // [grid*grid*Cout] f32 bits
                                     int npts, float bev, int gsz, int Cout) {
    int i = blockIdx.x * blockDim.x + threadIdx.x;
    if (i >= npts) return;
    float x = xyz[3 * i + 0], y = xyz[3 * i + 1], z = xyz[3 * i + 2];
    int xi = (int)floorf((x - PC_MIN) / bev); xi = xi < 0 ? 0 : (xi > gsz - 1 ? gsz - 1 : xi);
    int yi = (int)floorf((y - PC_MIN) / bev); yi = yi < 0 ? 0 : (yi > gsz - 1 ? gsz - 1 : yi);
    float cx = ((float)xi + 0.5f) * bev + PC_MIN;
    float cy = ((float)yi + 0.5f) * bev + PC_MIN;
    float f[8] = {x, y, z, x - cx, y - cy, z, ptf[2 * i], ptf[2 * i + 1]};
    unsigned int* cell = pooled + ((size_t)yi * gsz + xi) * Cout;
    for (int co = 0; co < Cout; ++co) {
        float h = bias[co];
#pragma unroll
        for (int j = 0; j < 8; ++j) h += f[j] * w[j * Cout + co];
        h = h * g[co] + beta[co];
        h = h > 0.0f ? h : 0.0f;
        atomicMax(cell + co, __float_as_uint(h));
    }
}

// ---------------------------------------------------------------------------
// Implicit-GEMM 3x3 conv (pad 1, stride 1/2), fused BN affine + residual +
// ReLU.  GEMM view: M = Hout*Wout pixels, N = Cout, K = 9*Cin.
//
// Tiling: each wave owns a 16(M) x (JT*16)(N) output tile as JT 16x16 WMMA
// accumulators (JT = 4/2/1 chosen at compile time from Cout, so the unrolled
// j-loop has NO guards and the accumulators stay pinned in registers).
// The shared 16x32 A chunk is staged into LDS once per block with
// GLOBAL_LOAD_ASYNC_TO_LDS_B128 (ASYNCcnt), double-buffered, synchronized
// with s_wait_asynccnt + workgroup split-barrier, boundary pixels zero-filled
// via ds_store_b128.  Each wave rebuilds its A fragment with 2x ds_load_b128
// in the ISA 16-bit A layout (lane half h: K = 8h..8h+7, 16+8h..23+8h).
// All 2*JT B-fragment b128 loads are issued up front (one clause) before the
// JT back-to-back v_wmma_f32_16x16x32_bf16, so load latency is hidden behind
// earlier WMMAs instead of a full s_wait before each one.  B addresses use a
// scalar incremental element offset (no per-iteration 64-bit chains).
//
// The only runtime guard ("does this wave own any N-tile", needed for
// Cout=32) is forced into an SGPR with readfirstlane so it lowers to a
// scalar branch: EXEC is all-ones at every WMMA as the ISA requires.
//
// Layouts:
//   in   : NHWC bf16  [H][W][Cin]        (Cin multiple of 32)
//   wrep : bf16       [tap][Cout][Cin]
//   out  : NHWC bf16  [Hout][Wout][Cout]
//   outf : NCHW f32   [Cout][Hout*Wout]  (optional, final network output)
// ---------------------------------------------------------------------------
#define WPB 4   // waves per block (blockDim.x = 128)

template <int JT>
__global__ void conv3x3_wmma_kernel(const bf16* __restrict__ in,
                                    const bf16* __restrict__ wrep,
                                    const float* __restrict__ gamma,
                                    const float* __restrict__ beta,
                                    const bf16* __restrict__ resid,
                                    bf16* __restrict__ out,
                                    float* __restrict__ outf,
                                    int H, int W, int Cin, int Cout,
                                    int Hout, int Wout, int stride, int relu) {
    __shared__ bf16 smem[2 * 16 * 32];          // double-buffered A chunk, 2 x 1KB

    const int tid   = threadIdx.x;
    const int lane  = tid & 31;
    const int wave  = tid >> 5;
    const int M     = Hout * Wout;
    const int mtile = blockIdx.x;

    const int half = lane >> 4;
    const int lm   = lane & 15;

    // JT interleaved N-subtiles per wave; only J=0 may be out of range
    // (Cout=32, JT=1, waves 2..3) -> single uniform scalar guard.
    int n0[JT];
    const bf16* bb[JT];
#pragma unroll
    for (int j = 0; j < JT; ++j) {
        n0[j] = (blockIdx.y * (JT * WPB) + j * WPB + wave) * 16;
        bb[j] = wrep + (size_t)((n0[j] < Cout ? n0[j] : 0) + lm) * Cin;
    }
    const int wact = __builtin_amdgcn_readfirstlane(n0[0] < Cout ? 1 : 0);

    // Staging geometry: threads 0..63 move the 16x32 chunk (4 x b128 per row).
    const int sr  = tid >> 2;                   // pixel row 0..15
    const int sp  = tid & 3;                    // 8-channel slice 0..3
    const int sm  = mtile * 16 + sr;
    const bool smv = (tid < 64) && (sm < M);
    int soy = 0, sox = 0;
    if (smv) { soy = sm / Wout; sox = sm - soy * Wout; }

    auto stage = [&](int tap, int ci0, int bufE) {
        if (tid < 64) {
            const int ky  = tap / 3;
            const int kx  = tap - 3 * ky;
            const int yin = soy * stride + ky - 1;
            const int xin = sox * stride + kx - 1;
            bf16* lp = &smem[bufE + sr * 32 + sp * 8];
            if (smv && (unsigned)yin < (unsigned)H && (unsigned)xin < (unsigned)W) {
                const bf16* gp = in + ((size_t)yin * W + xin) * Cin + ci0 + sp * 8;
                asm volatile("global_load_async_to_lds_b128 %0, %1, off"
                             :: "v"((unsigned)(uintptr_t)lp),
                                "v"((unsigned long long)(uintptr_t)gp)
                             : "memory");
            } else {
                v8bf z = {};
                *(v8bf*)lp = z;                 // ds_store_b128 of zeros (padding)
            }
        }
    };

    v8f acc[JT] = {};
    const int nit     = 9 * (Cin >> 5);
    const int tapstep = Cout * Cin - Cin + 32;  // B offset step on tap wrap
    int tap = 0, ci0 = 0, buf = 0, boff = 0;    // boff: scalar B element offset
    stage(0, 0, 0);
    for (int it = 0; it < nit; ++it) {
        // stage(it) complete (staging waves wait before the barrier) ...
        asm volatile("s_wait_asynccnt 0" ::: "memory");
        // ... and everyone is done consuming chunk it-1.
        __syncthreads();

        int ntap = tap, nci = ci0 + 32;
        if (nci == Cin) { nci = 0; ++ntap; }
        if (it + 1 < nit) stage(ntap, nci, (buf ^ 1) * 512);   // overlap with compute

        if (wact) {                             // scalar branch, EXEC untouched
            const int ab = buf * 512 + lm * 32 + 8 * half;
            v8bf alo = *(const v8bf*)&smem[ab];                // ds_load_b128
            v8bf ahi = *(const v8bf*)&smem[ab + 16];           // ds_load_b128
            v16bf a = __builtin_shufflevector(alo, ahi, 0, 1, 2, 3, 4, 5, 6, 7,
                                              8, 9, 10, 11, 12, 13, 14, 15);
            if (ci0 + 32 < Cin)
                __builtin_prefetch(bb[0] + boff + 8 * half + 32, 0, 0); // global_prefetch_b8
            // Issue ALL B-fragment loads first (single clause), then the WMMAs:
            // later WMMAs hide the latency of the earlier fragments' loads.
            v8bf blo[JT], bhi[JT];
#pragma unroll
            for (int j = 0; j < JT; ++j) {
                const bf16* bp = bb[j] + boff + 8 * half;
                blo[j] = *(const v8bf*)(bp);
                bhi[j] = *(const v8bf*)(bp + 16);
            }
#pragma unroll
            for (int j = 0; j < JT; ++j) {
                v16bf b = __builtin_shufflevector(blo[j], bhi[j], 0, 1, 2, 3, 4, 5, 6, 7,
                                                  8, 9, 10, 11, 12, 13, 14, 15);
                acc[j] = __builtin_amdgcn_wmma_f32_16x16x32_bf16(
                    /*neg_a=*/false, a, /*neg_b=*/false, b,
                    /*c_mod=*/(short)0, acc[j], /*reuse_a=*/false, /*reuse_b=*/false);
            }
        }
        boff += (nci != 0) ? 32 : tapstep;
        tap = ntap; ci0 = nci; buf ^= 1;
    }

    if (!wact) return;

    // Epilogue: BN affine + residual + ReLU; lane l holds rows r+8*half, col co.
#pragma unroll
    for (int j = 0; j < JT; ++j) {
        const int   co = n0[j] + lm;
        const float gc = gamma[co];
        const float bc = beta[co];
#pragma unroll
        for (int r = 0; r < 8; ++r) {
            int mm = mtile * 16 + r + 8 * half;
            if (mm >= M) continue;
            float v = acc[j][r] * gc + bc;
            if (resid) v += (float)resid[(size_t)mm * Cout + co];
            if (relu) v = v > 0.0f ? v : 0.0f;
            out[(size_t)mm * Cout + co] = (bf16)v;
            if (outf) outf[(size_t)co * M + mm] = v;
        }
    }
}

// ---------------------------------------------------------------------------
// Host orchestration
// ---------------------------------------------------------------------------
struct ConvP  { const float *w, *g, *b; };
struct BlockP { bool v; ConvP c0, c1, c2; };
struct StageP { bool down; ConvP dn; int nb; BlockP blk[6]; };
struct PPP    { const float *beta, *bias, *g, *w; };

static void conv_bn(const bf16* in, int H, int W, int Cin,
                    const ConvP& cv, int Cout, int stride,
                    const bf16* resid, bf16* out, float* outf,
                    int relu, bf16* wscratch, hipStream_t s) {
    int total = 9 * Cout * Cin;
    repack_w_kernel<<<(total + 255) / 256, 256, 0, s>>>(cv.w, wscratch, Cin, Cout, total);
    int Hout = H / stride, Wout = W / stride;   // stride-2 dims are all even here
    int M = Hout * Wout;
    int tiles = Cout >> 4;
    int JT = (tiles >= 4 * WPB) ? 4 : ((tiles >= 2 * WPB) ? 2 : 1);
    dim3 gsz((M + 15) / 16, (tiles + JT * WPB - 1) / (JT * WPB));
    switch (JT) {
    case 4:
        conv3x3_wmma_kernel<4><<<gsz, WPB * 32, 0, s>>>(in, wscratch, cv.g, cv.b, resid, out,
                                                        outf, H, W, Cin, Cout, Hout, Wout,
                                                        stride, relu);
        break;
    case 2:
        conv3x3_wmma_kernel<2><<<gsz, WPB * 32, 0, s>>>(in, wscratch, cv.g, cv.b, resid, out,
                                                        outf, H, W, Cin, Cout, Hout, Wout,
                                                        stride, relu);
        break;
    default:
        conv3x3_wmma_kernel<1><<<gsz, WPB * 32, 0, s>>>(in, wscratch, cv.g, cv.b, resid, out,
                                                        outf, H, W, Cin, Cout, Hout, Wout,
                                                        stride, relu);
        break;
    }
}

extern "C" void kernel_launch(void* const* d_in, const int* in_sizes, int n_in,
                              void* d_out, int out_size, void* d_ws, size_t ws_size,
                              hipStream_t stream) {
    (void)n_in; (void)out_size;
    const float* xyz = (const float*)d_in[0];
    const float* ptf = (const float*)d_in[2];
    const int npts = in_sizes[0] / 3;

    // ---- parse params (JAX pytree leaf order: sorted dict keys) ----
    int cur_in = 3;
    auto F = [&]() { return (const float*)d_in[cur_in++]; };
    auto readBlock = [&](bool v) {
        BlockP b{}; b.v = v;
        if (v) { b.c0.b = F(); b.c1.b = F(); b.c2.b = F();
                 b.c0.g = F(); b.c1.g = F(); b.c2.g = F();
                 b.c0.w = F(); b.c1.w = F(); b.c2.w = F(); }
        else   { b.c1.b = F(); b.c2.b = F();
                 b.c1.g = F(); b.c2.g = F();
                 b.c1.w = F(); b.c2.w = F(); }
        return b;
    };
    auto readStage = [&](int nb, bool v, bool down) {
        StageP s{}; s.nb = nb; s.down = down;
        for (int i = 0; i < nb; ++i) s.blk[i] = readBlock(v && i == 0);
        if (down) { s.dn.b = F(); s.dn.g = F(); s.dn.w = F(); }
        return s;
    };
    StageP c1_1 = readStage(3, true,  false);
    StageP c1_2 = readStage(3, true,  false);
    StageP c2_1 = readStage(4, false, true);
    StageP c2_2 = readStage(4, false, true);
    StageP c3_1 = readStage(6, false, true);
    StageP c3_2 = readStage(6, false, true);
    StageP c4_1 = readStage(3, false, true);
    StageP c4_2 = readStage(3, false, true);
    PPP pp1{F(), F(), F(), F()};
    PPP pp2{F(), F(), F(), F()};

    // ---- workspace layout ----
    // [0, 2MB)        : repacked weight scratch (max 9*256*256*2B = 1.18MB)
    // [2MB, 82MB)     : region 0 (also holds f32 pooled grid: 752^2*32*4 = 72.4MB)
    // [82MB, 122MB)   : region 1
    // [122MB, 162MB)  : region 2
    if (ws_size < (162ull << 20)) return;
    char* ws = (char*)d_ws;
    bf16* wS = (bf16*)ws;
    bf16* buf[3] = { (bf16*)(ws + (2ull << 20)),
                     (bf16*)(ws + (82ull << 20)),
                     (bf16*)(ws + (122ull << 20)) };

    StageP* br1[4] = {&c1_1, &c2_1, &c3_1, &c4_1}; const int dn1[4] = {0, 64, 128, 256};
    StageP* br2[4] = {&c1_2, &c2_2, &c3_2, &c4_2}; const int dn2[4] = {0, 128, 256, 256};

    auto runBranch = [&](const PPP& pp, float bev, int grid0, int Cpp,
                         StageP** stages, const int* dnC, float* finalOut) {
        // pillar scatter-max pool -> NHWC f32 in region 0, cast to bf16 in region 1
        size_t cells = (size_t)grid0 * grid0 * Cpp;
        fill_zero_kernel<<<2048, 256, 0, stream>>>((float*)buf[0], cells);
        pillar_encode_kernel<<<(npts + 255) / 256, 256, 0, stream>>>(
            xyz, ptf, pp.w, pp.bias, pp.g, pp.beta,
            (unsigned int*)buf[0], npts, bev, grid0, Cpp);
        cvt_f32_bf16_kernel<<<2048, 256, 0, stream>>>((const float*)buf[0], buf[1], cells);

        int cur = 1, C = Cpp, H = grid0, W = grid0;
        for (int si = 0; si < 4; ++si) {
            StageP& S = *stages[si];
            if (S.down) {
                int Cn = dnC[si];
                conv_bn(buf[cur], H, W, C, S.dn, Cn, 2, nullptr,
                        buf[(cur + 1) % 3], nullptr, 1, wS, stream);
                cur = (cur + 1) % 3; C = Cn; H /= 2; W /= 2;
            }
            for (int bi = 0; bi < S.nb; ++bi) {
                BlockP& B = S.blk[bi];
                bool last = (si == 3 && bi == S.nb - 1);
                if (B.v) {  // x0 = relu(bn(conv0(x)))
                    conv_bn(buf[cur], H, W, C, B.c0, C, 1, nullptr,
                            buf[(cur + 1) % 3], nullptr, 1, wS, stream);
                    cur = (cur + 1) % 3;
                }
                int a = cur, t = (cur + 1) % 3, o = (cur + 2) % 3;
                // h = relu(bn(conv1(x)));  out = relu(bn(conv2(h)) + x)
                conv_bn(buf[a], H, W, C, B.c1, C, 1, nullptr, buf[t], nullptr, 1, wS, stream);
                conv_bn(buf[t], H, W, C, B.c2, C, 1, buf[a], buf[o],
                        last ? finalOut : nullptr, 1, wS, stream);
                cur = o;
            }
        }
    };

    float* out_f32 = (float*)d_out;
    runBranch(pp1, 0.2f, 752, 32, br1, dn1, out_f32);                      // x_conv4 [256,94,94]
    runBranch(pp2, 0.4f, 376, 64, br2, dn2, out_f32 + 256 * 94 * 94);      // x_conv5 [256,47,47]
}
